// PMLP_GCN_79353815761143
// MI455X (gfx1250) — matrix operand order repaired and checked by reading, not compiled, not run on hardware
//
#include <hip/hip_runtime.h>
#include <hip/hip_bf16.h>

// ---------------------------------------------------------------------------
// Types for CDNA5 WMMA (wave32, 16x16x32 bf16 -> f32)
// ---------------------------------------------------------------------------
typedef __bf16 bf16;
typedef __attribute__((ext_vector_type(16))) __bf16 v16bf;
typedef __attribute__((ext_vector_type(8)))  __bf16 v8bf;
typedef __attribute__((ext_vector_type(8)))  float  v8f;

// Native convert: clang fptrunc f32->bf16 is round-to-nearest-even and lowers
// to v_cvt_pk_bf16_f32 on gfx1250 (vs ~5 VALU ops for the manual version).
__device__ __forceinline__ bf16 f2bf(float f) { return (bf16)f; }

// ---------------------------------------------------------------------------
// GEMM: C[M x N] = A[M x K] (f32, row major) * W[N x K]^T (f32, row major)
// N = NT*16.  One wave per 16-row M tile, NT accumulator tiles in registers.
// W staged once per block into LDS as bf16 [n][LDSP], zero-padded to Kpad.
// LDSP = Kpad + 24  -> row stride 12 dwords mod 64: conflict-free banks,
// rows stay 16B aligned for b128 LDS loads.
// All NT B-fragments are loaded before the WMMA group so a single
// s_wait_dscnt covers them and the NT WMMAs issue back-to-back.
// ---------------------------------------------------------------------------
template <int NT>
__global__ void gemm_wmma_bf16(const float* __restrict__ A,
                               const float* __restrict__ W,
                               float* __restrict__ C,
                               int M, int K, int Kpad) {
  extern __shared__ char smem_raw[];
  bf16* lds = (bf16*)smem_raw;
  const int N    = NT * 16;
  const int LDSP = Kpad + 24;

  // Cooperative stage: W (f32) -> LDS (bf16), zero-pad K..Kpad
  for (int idx = threadIdx.x; idx < N * Kpad; idx += blockDim.x) {
    int n = idx / Kpad;
    int k = idx - n * Kpad;
    lds[n * LDSP + k] = (k < K) ? f2bf(W[(long)n * K + k]) : f2bf(0.0f);
  }
  __syncthreads();

  const int wave = threadIdx.x >> 5;
  const int lane = threadIdx.x & 31;
  const int l    = lane & 15;   // 0..15
  const int hi   = lane >> 4;   // 0 or 1

  const int tileM = blockIdx.x * (blockDim.x >> 5) + wave;
  const int mb    = tileM * 16;
  if (mb >= M) return;

  int row = mb + l;
  if (row >= M) row = M - 1;    // clamp: OOB rows guarded at store
  const float* __restrict__ arow = A + (long)row * K;

  v8f acc[NT] = {};

  for (int kb = 0; kb < Kpad; kb += 32) {
    // ---- A fragment (16x32 bf16): lane<16 -> K {kb..+15},
    //                                lane>=16 -> K {kb+16..+31}
    v16bf av;
    const int k0 = kb + 16 * hi;
    const int k1 = kb + 16 * hi + 8;
    if (kb + 32 <= K) {
#pragma unroll
      for (int j = 0; j < 8; ++j) {
        av[j]     = f2bf(arow[k0 + j]);
        av[8 + j] = f2bf(arow[k1 + j]);
      }
    } else {  // K tail (e.g. K=500, Kpad=512): guarded, zero-fill
#pragma unroll
      for (int j = 0; j < 8; ++j) {
        av[j]     = (k0 + j < K) ? f2bf(arow[k0 + j]) : f2bf(0.0f);
        av[8 + j] = (k1 + j < K) ? f2bf(arow[k1 + j]) : f2bf(0.0f);
      }
    }

    // ---- Load ALL B fragments first (one dscnt wait covers all),
    //      then issue the NT WMMAs back-to-back on the matrix pipe.
    v16bf bfr[NT];
#pragma unroll
    for (int t = 0; t < NT; ++t) {
      const bf16* bp = &lds[(t * 16 + l) * LDSP + kb + 16 * hi];
      v8bf blo = *(const v8bf*)(bp);
      v8bf bhi = *(const v8bf*)(bp + 8);
#pragma unroll
      for (int j = 0; j < 8; ++j) { bfr[t][j] = blo[j]; bfr[t][8 + j] = bhi[j]; }
    }
#pragma unroll
    for (int t = 0; t < NT; ++t) {
      acc[t] = __builtin_amdgcn_wmma_f32_16x16x32_bf16(
          false, av, false, bfr[t], (short)0, acc[t], false, false);
    }
  }

  // ---- Store D: VGPR r -> row (mb + r + 8*hi), col (nt*16 + l)
#pragma unroll
  for (int t = 0; t < NT; ++t) {
#pragma unroll
    for (int r = 0; r < 8; ++r) {
      const int orow = mb + r + 8 * hi;
      if (orow < M) C[(long)orow * N + t * 16 + l] = acc[t][r];
    }
  }
}

// ---------------------------------------------------------------------------
// Graph aggregation + batchnorm + softmax helpers
// ---------------------------------------------------------------------------
__global__ void zero_f32(float* __restrict__ p, int n) {
  int i = blockIdx.x * blockDim.x + threadIdx.x;
  if (i < n) p[i] = 0.0f;
}

__global__ void deg_count(const int* __restrict__ src, const int* __restrict__ dst,
                          float* __restrict__ deg, int E) {
  int e = blockIdx.x * blockDim.x + threadIdx.x;
  if (e >= E) return;
  int s = src[e], d = dst[e];
  if (s != d) atomicAdd(&deg[d], 1.0f);
}

__global__ void deg_final(const float* __restrict__ deg,
                          float* __restrict__ dinv, float* __restrict__ invdeg, int n) {
  int i = blockIdx.x * blockDim.x + threadIdx.x;
  if (i >= n) return;
  float d = deg[i] + 1.0f;        // +1 for added self loop
  dinv[i]   = rsqrtf(d);
  invdeg[i] = 1.0f / d;
}

// out[i,f] = invdeg[i] * h[i,f] + bias[f]   (self-loop term + bias)
__global__ void agg_init(const float* __restrict__ h, const float* __restrict__ invdeg,
                         const float* __restrict__ bias, float* __restrict__ out,
                         int n, int F) {
  int i = blockIdx.x * blockDim.x + threadIdx.x;
  if (i >= n * F) return;
  int r = i / F, c = i - r * F;
  out[i] = invdeg[r] * h[i] + bias[c];
}

// one wave per edge; lane handles F/32 consecutive features
template <int F>
__global__ void agg_edges(const float* __restrict__ h, float* __restrict__ out,
                          const int* __restrict__ src, const int* __restrict__ dst,
                          const float* __restrict__ dinv, int E) {
  int e = (blockIdx.x * blockDim.x + threadIdx.x) >> 5;
  if (e >= E) return;
  int lane = threadIdx.x & 31;
  int s = src[e], d = dst[e];
  if (s == d) return;                       // removed self loops
  float w = dinv[s] * dinv[d];
  constexpr int VPT = F / 32;
  const float* __restrict__ hs = h   + (long)s * F + lane * VPT;
  float* __restrict__       od = out + (long)d * F + lane * VPT;
#pragma unroll
  for (int j = 0; j < VPT; ++j) atomicAdd(&od[j], w * hs[j]);
}

// column sum / sumsq partials (consecutive threads -> consecutive cols)
__global__ void col_stats(const float* __restrict__ h, float* __restrict__ stats,
                          int n, int F, int rowsPerThread) {
  int t = blockIdx.x * blockDim.x + threadIdx.x;
  int c = t % F;
  int r0 = (t / F) * rowsPerThread;
  if (r0 >= n) return;
  int r1 = min(n, r0 + rowsPerThread);
  float s = 0.0f, sq = 0.0f;
  for (int r = r0; r < r1; ++r) {
    float v = h[(long)r * F + c];
    s += v; sq += v * v;
  }
  atomicAdd(&stats[c], s);
  atomicAdd(&stats[F + c], sq);
}

__global__ void bn_relu(const float* __restrict__ h, const float* __restrict__ stats,
                        float* __restrict__ out, int n, int F) {
  int i = blockIdx.x * blockDim.x + threadIdx.x;
  if (i >= n * F) return;
  int c = i % F;
  float invn = 1.0f / (float)n;
  float mean = stats[c] * invn;
  float var  = stats[F + c] * invn - mean * mean;   // biased var
  float v = (h[i] - mean) * rsqrtf(var + 1e-5f);
  out[i] = v > 0.0f ? v : 0.0f;
}

// one wave per row of 64 logits (2 per lane), __shfl_xor wave32 reductions
__global__ void log_softmax64(const float* __restrict__ h, float* __restrict__ out, int n) {
  int row = (blockIdx.x * blockDim.x + threadIdx.x) >> 5;
  if (row >= n) return;
  int lane = threadIdx.x & 31;
  const float* hr = h + (long)row * 64;
  float a = hr[lane], b = hr[lane + 32];
  float m = fmaxf(a, b);
#pragma unroll
  for (int off = 16; off > 0; off >>= 1) m = fmaxf(m, __shfl_xor(m, off, 32));
  float s = __expf(a - m) + __expf(b - m);
#pragma unroll
  for (int off = 16; off > 0; off >>= 1) s += __shfl_xor(s, off, 32);
  float ls = __logf(s);
  out[(long)row * 64 + lane]      = a - m - ls;
  out[(long)row * 64 + lane + 32] = b - m - ls;
}

// ---------------------------------------------------------------------------
// Driver
// ---------------------------------------------------------------------------
static inline int cdiv(int a, int b) { return (a + b - 1) / b; }

extern "C" void kernel_launch(void* const* d_in, const int* in_sizes, int n_in,
                              void* d_out, int out_size, void* d_ws, size_t ws_size,
                              hipStream_t stream) {
  const float* x  = (const float*)d_in[0];
  const int*   ei = (const int*)  d_in[1];
  const float* W0 = (const float*)d_in[2];
  const float* b0 = (const float*)d_in[3];
  const float* W1 = (const float*)d_in[4];
  const float* b1 = (const float*)d_in[5];
  const float* W2 = (const float*)d_in[6];
  const float* b2 = (const float*)d_in[7];

  const int HID = in_sizes[3];              // 128
  const int OUT = in_sizes[7];              // 64
  const int IN  = in_sizes[2] / HID;        // 500
  const int Nn  = in_sizes[0] / IN;         // 50000
  const int E   = in_sizes[1] / 2;          // 640000
  const int* src = ei;
  const int* dst = ei + E;

  // workspace layout (f32)
  float* ws     = (float*)d_ws;
  float* deg    = ws;                       // Nn
  float* dinv   = deg + Nn;                 // Nn
  float* invdeg = dinv + Nn;                // Nn
  float* stats  = invdeg + Nn;              // 2*HID
  float* H      = stats + 2 * HID;          // Nn*HID
  float* G      = H + (size_t)Nn * HID;     // Nn*HID

  const int T = 256;

  // ---- degrees / normalization coefficients
  zero_f32 <<<cdiv(Nn, T), T, 0, stream>>>(deg, Nn);
  deg_count<<<cdiv(E, T),  T, 0, stream>>>(src, dst, deg, E);
  deg_final<<<cdiv(Nn, T), T, 0, stream>>>(deg, dinv, invdeg, Nn);

  // ---- layer 0: H = x @ W0^T  (K=500 -> Kpad=512, N=128)
  {
    int Kpad = (IN + 31) & ~31;
    int blocks = cdiv(cdiv(Nn, 16), 8);
    size_t lds = (size_t)(8 * 16) * (Kpad + 24) * sizeof(bf16);
    gemm_wmma_bf16<8><<<blocks, T, lds, stream>>>(x, W0, H, Nn, IN, Kpad);
  }
  agg_init <<<cdiv(Nn * HID, T), T, 0, stream>>>(H, invdeg, b0, G, Nn, HID);
  agg_edges<128><<<cdiv(E * 32, T), T, 0, stream>>>(H, G, src, dst, dinv, E);
  zero_f32 <<<1, T, 0, stream>>>(stats, 2 * HID);
  col_stats<<<cdiv(cdiv(Nn, 64) * HID, T), T, 0, stream>>>(G, stats, Nn, HID, 64);
  bn_relu  <<<cdiv(Nn * HID, T), T, 0, stream>>>(G, stats, H, Nn, HID);

  // ---- layer 1: G = H @ W1^T  (K=128, N=128)
  {
    int Kpad = (HID + 31) & ~31;
    int blocks = cdiv(cdiv(Nn, 16), 8);
    size_t lds = (size_t)(8 * 16) * (Kpad + 24) * sizeof(bf16);
    gemm_wmma_bf16<8><<<blocks, T, lds, stream>>>(H, W1, G, Nn, HID, Kpad);
  }
  agg_init <<<cdiv(Nn * HID, T), T, 0, stream>>>(G, invdeg, b1, H, Nn, HID);
  agg_edges<128><<<cdiv(E * 32, T), T, 0, stream>>>(G, H, src, dst, dinv, E);
  zero_f32 <<<1, T, 0, stream>>>(stats, 2 * HID);
  col_stats<<<cdiv(cdiv(Nn, 64) * HID, T), T, 0, stream>>>(H, stats, Nn, HID, 64);
  bn_relu  <<<cdiv(Nn * HID, T), T, 0, stream>>>(H, stats, G, Nn, HID);

  // ---- classifier: H(:, :64) = G @ W2^T  (K=128, N=64)
  {
    int Kpad = (HID + 31) & ~31;
    int blocks = cdiv(cdiv(Nn, 16), 8);
    size_t lds = (size_t)(4 * 16) * (Kpad + 24) * sizeof(bf16);
    gemm_wmma_bf16<4><<<blocks, T, lds, stream>>>(G, W2, H, Nn, HID, Kpad);
  }
  agg_init <<<cdiv(Nn * OUT, T), T, 0, stream>>>(H, invdeg, b2, G, Nn, OUT);
  agg_edges<64><<<cdiv(E * 32, T), T, 0, stream>>>(H, G, src, dst, dinv, E);

  // ---- log_softmax -> d_out
  log_softmax64<<<cdiv(Nn * 32, T), T, 0, stream>>>(G, (float*)d_out, Nn);
}
